// BiLstm_Crf_21002390077888
// MI455X (gfx1250) — compile-verified, matrix operand order checked by reading, not run on hardware
//
#include <hip/hip_runtime.h>
#include <hip/hip_bf16.h>
#include <math.h>

typedef __attribute__((ext_vector_type(16))) __bf16 v16bf;
typedef __attribute__((ext_vector_type(8)))  float  v8f;
typedef __attribute__((ext_vector_type(8)))  int    v8i;

#define T_LEN  512
#define B_SZ   64
#define E_DIM  300
#define E_PAD  320
#define H_DIM  256
#define G_DIM  1024   /* 4H */
#define K_TAGS 20

// ---------------------------------------------------------------- helpers ---

__device__ __forceinline__ unsigned short f2bf(float f) {
  union { float f; unsigned u; } x; x.f = f;
  unsigned r = (x.u + 0x7FFFu + ((x.u >> 16) & 1u)) >> 16;  // RNE
  return (unsigned short)r;
}

__device__ __forceinline__ float sigmoidf_(float x) {
  return 1.0f / (1.0f + __expf(-x));
}

__device__ __forceinline__ v8f zero8() {
  v8f z = {0.f, 0.f, 0.f, 0.f, 0.f, 0.f, 0.f, 0.f};
  return z;
}

// A-fragment: 16x32 bf16 tile from a row-major bf16 matrix (stride in elems).
// Per CDNA5 ISA 16-bit A layout: lane = half*16 + m;
// VGPR v holds K = (v>=4?16:0) + (v&3)*2 + half*8 + {0,1}.
__device__ __forceinline__ v16bf load_a_frag(const unsigned short* base,
                                             int stride, int kbase, int lane) {
  const int m = lane & 15, half = lane >> 4;
  union { unsigned u[8]; v16bf v; } r;
  const unsigned short* p = base + m * stride + kbase + half * 8;
#pragma unroll
  for (int v = 0; v < 8; ++v) {
    int k = ((v & 4) << 2) + ((v & 3) << 1);
    r.u[v] = *(const unsigned*)(p + k);   // 2 consecutive bf16 per dword
  }
  return r.v;
}

// B-fragment at a constant element offset from a per-wave base pointer
// (base already includes ntile-base*512 + lane*16). Offsets stay compile-time
// constants so they fold into the 24-bit instruction IOFFSET.
__device__ __forceinline__ v16bf load_b_at(const unsigned short* p, int elem_off) {
  union { v8i i; v16bf v; } r;
  r.i = *(const v8i*)(p + elem_off);
  return r.v;
}

__device__ __forceinline__ v8f wmma_bf16(v16bf a, v16bf b, v8f c) {
  // (neg_a, A, neg_b, B, c_mod, C, reuse_a, reuse_b)
  return __builtin_amdgcn_wmma_f32_16x16x32_bf16(false, a, false, b,
                                                 (short)0, c, false, false);
}

// ------------------------------------------------------------ pack kernels ---

// Shuffle W (nrows x ncols, row-major f32) into B-fragment layout:
// B[k][n] = W[n][k], tiles (kt, nt), frag[((kt*ntiles+nt)*32 + lane)*16 + e].
__global__ void pack_b_kernel(const float* __restrict__ src,
                              unsigned short* __restrict__ dst,
                              int ktiles, int ntiles, int nrows, int ncols) {
  int idx = blockIdx.x * blockDim.x + threadIdx.x;
  int total = ktiles * ntiles * 512;
  if (idx >= total) return;
  int e = idx & 15;
  int lane = (idx >> 4) & 31;
  int tile = idx >> 9;
  int nt = tile % ntiles, kt = tile / ntiles;
  int half = lane >> 4, nl = lane & 15;
  int v = e >> 1, j = e & 1;
  int k = kt * 32 + ((v & 4) << 2) + ((v & 3) << 1) + half * 8 + j;
  int n = nt * 16 + nl;
  float val = 0.0f;
  if (n < nrows && k < ncols) val = src[(size_t)n * ncols + k];
  dst[idx] = f2bf(val);
}

__global__ void bias_kernel(const float* __restrict__ bih,
                            const float* __restrict__ bhh,
                            float* __restrict__ bias) {
  int i = blockIdx.x * blockDim.x + threadIdx.x;
  if (i < G_DIM) bias[i] = bih[i] + bhh[i];
}

// Embedding gather -> bf16, E padded 300 -> 320 with zeros.
__global__ void embed_kernel(const int* __restrict__ inp,
                             const float* __restrict__ emb,
                             unsigned short* __restrict__ xbf) {
  long long idx = (long long)blockIdx.x * blockDim.x + threadIdx.x;
  const long long total = (long long)T_LEN * B_SZ * E_PAD;
  if (idx >= total) return;
  int e  = (int)(idx % E_PAD);
  int tb = (int)(idx / E_PAD);
  float val = 0.0f;
  if (e < E_DIM) {
    int tok = inp[tb];
    val = emb[(long long)tok * E_DIM + e];
  }
  xbf[idx] = f2bf(val);
}

// --------------------------------------------------------------- LSTM core ---
// grid = 4 blocks (16 batch rows each, independent recurrences),
// block = 512 threads = 16 waves. Wave w owns hidden columns [w*16, w*16+16)
// and accumulates its i/f/g/o gate tiles in 4 v8f registers; cell state c
// lives in registers, h round-trips through an 8KB bf16 LDS tile.
// Per-wave weight fragment base pointers are hoisted; a laundered zero index
// is folded in once per timestep (defeats LICM without losing pointer
// provenance), and all fragment offsets are compile-time constants that fold
// into the instruction IOFFSET -> ~1 address add per loop body.
__global__ void __launch_bounds__(512) lstm_kernel(
    const unsigned short* __restrict__ xbf,    // (T,B,E_PAD) bf16
    const unsigned short* __restrict__ wihT,   // B-frags: 10 ktiles x 64 ntiles
    const unsigned short* __restrict__ whhT,   // B-frags:  8 ktiles x 64 ntiles
    const float* __restrict__ bias,            // b_ih + b_hh (1024)
    unsigned short* __restrict__ outs) {       // (T,B,H) bf16
  __shared__ unsigned short s_h[16 * H_DIM];

  const int tid  = threadIdx.x;
  const int w    = tid >> 5;
  const int lane = tid & 31;
  const int half = lane >> 4, nl = lane & 15;
  const int bbase = blockIdx.x * 16;
  const int col   = w * 16 + nl;               // hidden column owned

  for (int i = tid; i < 16 * H_DIM; i += blockDim.x) s_h[i] = 0;
  __syncthreads();

  const float bI = bias[0 * H_DIM + col];
  const float bF = bias[1 * H_DIM + col];
  const float bG = bias[2 * H_DIM + col];
  const float bO = bias[3 * H_DIM + col];

  // Per-wave fragment bases: tile (kt, gate*16+w) lives at constant offset
  // kt*32768 + gate*8192 elements from here.
  const unsigned short* wih_w = wihT + (size_t)(w * 512 + lane * 16);
  const unsigned short* whh_w = whhT + (size_t)(w * 512 + lane * 16);

  float creg[8];
#pragma unroll
  for (int r = 0; r < 8; ++r) creg[r] = 0.0f;

  for (int t = 0; t < T_LEN; ++t) {
    // Laundered zero: opaque per-iteration value, always 0 at runtime.
    int lk = 0;
    asm volatile("" : "+s"(lk));
    const unsigned short* wih_t = wih_w + lk;
    const unsigned short* whh_t = whh_w + lk;

    v8f aI = zero8(), aF = zero8(), aG = zero8(), aO = zero8();
    const unsigned short* xrow = xbf + ((size_t)t * B_SZ + bbase) * E_PAD;
    __builtin_prefetch(xrow + (size_t)B_SZ * E_PAD, 0, 1);  // next timestep

    // gates += x_t @ w_ih^T   (K = 320)
#pragma unroll 2
    for (int kt = 0; kt < E_PAD / 32; ++kt) {
      v16bf a = load_a_frag(xrow, E_PAD, kt * 32, lane);
      aI = wmma_bf16(a, load_b_at(wih_t, kt * 32768 + 0 * 8192), aI);
      aF = wmma_bf16(a, load_b_at(wih_t, kt * 32768 + 1 * 8192), aF);
      aG = wmma_bf16(a, load_b_at(wih_t, kt * 32768 + 2 * 8192), aG);
      aO = wmma_bf16(a, load_b_at(wih_t, kt * 32768 + 3 * 8192), aO);
    }
    // gates += h_{t-1} @ w_hh^T   (K = 256), h read from LDS
#pragma unroll 2
    for (int kt = 0; kt < H_DIM / 32; ++kt) {
      v16bf a = load_a_frag(s_h, H_DIM, kt * 32, lane);
      aI = wmma_bf16(a, load_b_at(whh_t, kt * 32768 + 0 * 8192), aI);
      aF = wmma_bf16(a, load_b_at(whh_t, kt * 32768 + 1 * 8192), aF);
      aG = wmma_bf16(a, load_b_at(whh_t, kt * 32768 + 2 * 8192), aG);
      aO = wmma_bf16(a, load_b_at(whh_t, kt * 32768 + 3 * 8192), aO);
    }
    __syncthreads();   // everyone done reading s_h before it is overwritten

#pragma unroll
    for (int r = 0; r < 8; ++r) {
      float iv = sigmoidf_(aI[r] + bI);
      float fv = sigmoidf_(aF[r] + bF);
      float gv = tanhf(aG[r] + bG);
      float ov = sigmoidf_(aO[r] + bO);
      creg[r] = fv * creg[r] + iv * gv;
      float hv = ov * tanhf(creg[r]);
      unsigned short hb = f2bf(hv);
      int m = r + 8 * half;                   // C/D layout: VGPR r, half
      s_h[m * H_DIM + col] = hb;
      outs[((size_t)t * B_SZ + bbase + m) * H_DIM + col] = hb;
    }
    __syncthreads();
  }
}

// ---------------------------------------------------- emissions = h @ W^T ---
__global__ void __launch_bounds__(256) emis_kernel(
    const unsigned short* __restrict__ outs,   // (T*B, H) bf16
    const unsigned short* __restrict__ wlinT,  // B-frags: 8 ktiles x 2 ntiles
    const float* __restrict__ blin,            // (20)
    float* __restrict__ emis) {                // (T*B, 20) f32
  const int lane = threadIdx.x & 31;
  const int wv   = threadIdx.x >> 5;
  const int mtile = blockIdx.x * 8 + wv;       // 0..2047
  const int half = lane >> 4, nl = lane & 15;

  v8f a0 = zero8(), a1 = zero8();
  const unsigned short* hrow = outs + (size_t)mtile * 16 * H_DIM;
  const unsigned short* wlin_w = wlinT + (size_t)(lane * 16);
#pragma unroll 2
  for (int kt = 0; kt < H_DIM / 32; ++kt) {
    v16bf a = load_a_frag(hrow, H_DIM, kt * 32, lane);
    a0 = wmma_bf16(a, load_b_at(wlin_w, (kt * 2 + 0) * 512), a0);
    a1 = wmma_bf16(a, load_b_at(wlin_w, (kt * 2 + 1) * 512), a1);
  }
  float b0 = blin[nl];
#pragma unroll
  for (int r = 0; r < 8; ++r) {
    int row = mtile * 16 + r + 8 * half;
    emis[(size_t)row * K_TAGS + nl] = a0[r] + b0;
  }
  if (nl < K_TAGS - 16) {                      // columns 16..19 of padded tile
    float b1 = blin[16 + nl];
#pragma unroll
    for (int r = 0; r < 8; ++r) {
      int row = mtile * 16 + r + 8 * half;
      emis[(size_t)row * K_TAGS + 16 + nl] = a1[r] + b1;
    }
  }
}

// ------------------------------------------------------------- CRF forward ---
__global__ void __launch_bounds__(512) crf_kernel(
    const float* __restrict__ emis,     // (T,B,K)
    const int* __restrict__ labels,     // (T,B)
    const float* __restrict__ start_t,
    const float* __restrict__ end_t,
    const float* __restrict__ trans,    // (K,K)
    float* __restrict__ out) {
  __shared__ float alpha[2][B_SZ * K_TAGS];
  __shared__ float s_trans[K_TAGS * K_TAGS];
  __shared__ float red[B_SZ];
  const int tid = threadIdx.x;

  for (int i = tid; i < K_TAGS * K_TAGS; i += blockDim.x) s_trans[i] = trans[i];
  for (int i = tid; i < B_SZ * K_TAGS; i += blockDim.x) {
    int b = i / K_TAGS, k = i % K_TAGS;
    alpha[0][i] = start_t[k] + emis[(size_t)b * K_TAGS + k];
  }
  float score = 0.0f;
  if (tid < B_SZ) {
    int tag0 = labels[tid];
    score = start_t[tag0] + emis[(size_t)tid * K_TAGS + tag0];
  }

  for (int t = 1; t < T_LEN; ++t) {
    __syncthreads();
    const int cur = t & 1, prev = cur ^ 1;
    const float* em_t = emis + (size_t)t * B_SZ * K_TAGS;
    for (int i = tid; i < B_SZ * K_TAGS; i += blockDim.x) {
      int b = i / K_TAGS, kp = i % K_TAGS;
      float m = -INFINITY;
#pragma unroll
      for (int k = 0; k < K_TAGS; ++k)
        m = fmaxf(m, alpha[prev][b * K_TAGS + k] + s_trans[k * K_TAGS + kp]);
      float s = 0.0f;
#pragma unroll
      for (int k = 0; k < K_TAGS; ++k)
        s += __expf(alpha[prev][b * K_TAGS + k] + s_trans[k * K_TAGS + kp] - m);
      alpha[cur][i] = m + __logf(s) + em_t[b * K_TAGS + kp];
    }
    if (tid < B_SZ) {
      int tp = labels[(t - 1) * B_SZ + tid];
      int tc = labels[t * B_SZ + tid];
      score += s_trans[tp * K_TAGS + tc] + em_t[tid * K_TAGS + tc];
    }
  }
  __syncthreads();

  if (tid < B_SZ) {
    const float* arow = &alpha[(T_LEN - 1) & 1][tid * K_TAGS];
    float m = -INFINITY;
#pragma unroll
    for (int k = 0; k < K_TAGS; ++k) m = fmaxf(m, arow[k] + end_t[k]);
    float s = 0.0f;
#pragma unroll
    for (int k = 0; k < K_TAGS; ++k) s += __expf(arow[k] + end_t[k] - m);
    float logZ = m + __logf(s);
    int tlast = labels[(T_LEN - 1) * B_SZ + tid];
    red[tid] = -(score + end_t[tlast] - logZ);
  }
  __syncthreads();
  if (tid == 0) {
    float acc = 0.0f;
    for (int b = 0; b < B_SZ; ++b) acc += red[b];
    out[0] = acc;
  }
}

// -------------------------------------------------------------- launcher ----

extern "C" void kernel_launch(void* const* d_in, const int* in_sizes, int n_in,
                              void* d_out, int out_size, void* d_ws, size_t ws_size,
                              hipStream_t stream) {
  const int*   inp     = (const int*)  d_in[0];
  const int*   labels  = (const int*)  d_in[1];
  const float* emb     = (const float*)d_in[2];
  const float* w_ih    = (const float*)d_in[3];
  const float* w_hh    = (const float*)d_in[4];
  const float* b_ih    = (const float*)d_in[5];
  const float* b_hh    = (const float*)d_in[6];
  const float* W_lin   = (const float*)d_in[7];
  const float* b_lin   = (const float*)d_in[8];
  const float* start_t = (const float*)d_in[9];
  const float* end_t   = (const float*)d_in[10];
  const float* trans   = (const float*)d_in[11];
  float* out = (float*)d_out;

  char* ws = (char*)d_ws;
  size_t off = 0;
  auto carve = [&](size_t bytes) -> char* {
    char* p = ws + off;
    off = (off + bytes + 255) & ~(size_t)255;
    return p;
  };

  unsigned short* wihT  = (unsigned short*)carve((size_t)10 * 64 * 512 * 2);
  unsigned short* whhT  = (unsigned short*)carve((size_t)8  * 64 * 512 * 2);
  unsigned short* wlinT = (unsigned short*)carve((size_t)8  * 2  * 512 * 2);
  float*          bias  = (float*)         carve((size_t)G_DIM * 4);
  unsigned short* xbf   = (unsigned short*)carve((size_t)T_LEN * B_SZ * E_PAD * 2);
  unsigned short* outs  = (unsigned short*)carve((size_t)T_LEN * B_SZ * H_DIM * 2);
  float*          emis  = (float*)         carve((size_t)T_LEN * B_SZ * K_TAGS * 4);

  // weight shuffles into WMMA B-fragment layout (bf16)
  pack_b_kernel<<<(10 * 64 * 512 + 255) / 256, 256, 0, stream>>>(
      w_ih, wihT, 10, 64, G_DIM, E_DIM);
  pack_b_kernel<<<(8 * 64 * 512 + 255) / 256, 256, 0, stream>>>(
      w_hh, whhT, 8, 64, G_DIM, H_DIM);
  pack_b_kernel<<<(8 * 2 * 512 + 255) / 256, 256, 0, stream>>>(
      W_lin, wlinT, 8, 2, K_TAGS, H_DIM);
  bias_kernel<<<(G_DIM + 255) / 256, 256, 0, stream>>>(b_ih, b_hh, bias);

  // embedding gather -> bf16 padded
  embed_kernel<<<((size_t)T_LEN * B_SZ * E_PAD + 255) / 256, 256, 0, stream>>>(
      inp, emb, xbf);

  // recurrent LSTM: 4 independent batch groups, WMMA per step
  lstm_kernel<<<4, 512, 0, stream>>>(xbf, wihT, whhT, bias, outs);

  // emissions GEMM
  emis_kernel<<<256, 256, 0, stream>>>(outs, wlinT, b_lin, emis);

  // CRF gold score + forward algorithm + reduction
  crf_kernel<<<1, 512, 0, stream>>>(emis, labels, start_t, end_t, trans, out);
}
